// slot_dec_14645838479776
// MI455X (gfx1250) — compile-verified, matrix operand order checked by reading, not run on hardware
//
#include <hip/hip_runtime.h>
#include <hip/hip_bf16.h>
#include <stdint.h>

// Problem sizes (fixed by the reference)
#define H    1024
#define L    4096
#define G4H  4096   // 4*H
#define LBL  128

typedef unsigned short bf16_t;
typedef __attribute__((ext_vector_type(16))) __bf16 v16bf;
typedef __attribute__((ext_vector_type(8)))  float  v8f;

// ---------------- helpers ----------------
__device__ __forceinline__ bf16_t f2bf(float f) {
    union { float f; uint32_t u; } v; v.f = f;
    uint32_t r = (v.u + 0x7FFFu + ((v.u >> 16) & 1u)) >> 16;   // RNE
    return (bf16_t)r;
}
__device__ __forceinline__ float bf2f_lo(uint32_t u) {
    union { uint32_t u; float f; } v; v.u = u << 16; return v.f;
}
__device__ __forceinline__ float bf2f_hi(uint32_t u) {
    union { uint32_t u; float f; } v; v.u = u & 0xFFFF0000u; return v.f;
}
__device__ __forceinline__ float sigf(float x) { return 1.0f / (1.0f + __expf(-x)); }

// ---------------- prep kernels ----------------
__global__ void k_init(float* state, int nstate, unsigned* cnt) {
    int i = blockIdx.x * blockDim.x + threadIdx.x;
    if (i == 0) *cnt = 0u;
    if (i < nstate) state[i] = 0.0f;
}

// dst[r*cols + c] = bf16(src[r*srcStride + c])  (cols is a power of two)
__global__ void k_cvt_bf16(const float* __restrict__ src, size_t srcStride,
                           bf16_t* __restrict__ dst, int rows, int cols) {
    size_t n = (size_t)rows * (size_t)cols;
    size_t stride = (size_t)gridDim.x * blockDim.x;
    for (size_t i = (size_t)blockIdx.x * blockDim.x + threadIdx.x; i < n; i += stride) {
        size_t r = i / (size_t)cols, c = i - r * (size_t)cols;
        dst[i] = f2bf(src[r * srcStride + c]);
    }
}

// X4[t][0:2048] = x[t], X4[t][2048:4096] = hi[t]   (bf16)
__global__ void k_pack_x(const float* __restrict__ x, const float* __restrict__ hin,
                         bf16_t* __restrict__ X4) {
    size_t n = (size_t)L * G4H;
    size_t stride = (size_t)gridDim.x * blockDim.x;
    for (size_t i = (size_t)blockIdx.x * blockDim.x + threadIdx.x; i < n; i += stride) {
        size_t t = i >> 12, c = i & (G4H - 1);
        float v = (c < 2 * H) ? x[t * (2 * H) + c] : hin[t * (2 * H) + (c - 2 * H)];
        X4[i] = f2bf(v);
    }
}

// ---------------- WMMA GEMM:  C[M,N] = A[M,K](bf16) * B[N,K]^T(bf16) + bias[N] ----------------
// Block tile 128(M) x 64(N), K-step 32. 8 waves/block; wave w owns rows [16w,16w+16),
// 4 accumulator tiles across N. Emits v_wmma_f32_16x16x32_bf16.
#define BM 128
#define BN 64
#define BK 32

__global__ __launch_bounds__(256) void gemm_bf16_wmma(
    const bf16_t* __restrict__ A, const bf16_t* __restrict__ B,
    const float* __restrict__ bias, float* __restrict__ C,
    int M, int N, int K)
{
    __shared__ __align__(16) bf16_t As[BM][BK];
    __shared__ __align__(16) bf16_t Bs[BN][BK];

    const int tid  = threadIdx.x;
    const int wave = tid >> 5;
    const int lane = tid & 31;
    const int ln16 = lane & 15;
    const int hseg = lane >> 4;              // 0 or 1 (half-wave)
    const int bm = blockIdx.y * BM;
    const int bn = blockIdx.x * BN;

    v8f acc[4] = {};

    for (int k0 = 0; k0 < K; k0 += BK) {
        // Stage A tile (128x32 bf16 = 8KB): each thread moves 16 bf16 (two b128s)
        {
            int row = tid >> 1, half = tid & 1;
            const bf16_t* src = A + (size_t)(bm + row) * K + k0 + half * 16;
            uint4 q0 = ((const uint4*)src)[0];
            uint4 q1 = ((const uint4*)src)[1];
            *(uint4*)&As[row][half * 16 + 0] = q0;
            *(uint4*)&As[row][half * 16 + 8] = q1;
        }
        // Stage B tile (64x32 bf16 = 4KB)
        if (tid < 128) {
            int row = tid >> 1, half = tid & 1;
            const bf16_t* src = B + (size_t)(bn + row) * K + k0 + half * 16;
            uint4 q0 = ((const uint4*)src)[0];
            uint4 q1 = ((const uint4*)src)[1];
            *(uint4*)&Bs[row][half * 16 + 0] = q0;
            *(uint4*)&Bs[row][half * 16 + 8] = q1;
        }
        __syncthreads();

        // Per-lane fragment layout for 16-bit A 16x32 (ISA 7.12.2):
        // lanes 0-15 hold K = kb..kb+7 and 16+kb..23+kb with kb=0; lanes 16-31: kb=8.
        union Frag { v16bf v; uint4 q[2]; };
        const int kb = hseg * 8;

        Frag af;
        af.q[0] = *(const uint4*)&As[wave * 16 + ln16][kb];
        af.q[1] = *(const uint4*)&As[wave * 16 + ln16][16 + kb];

        Frag bfrag[4];
#pragma unroll
        for (int nt = 0; nt < 4; ++nt) {
            bfrag[nt].q[0] = *(const uint4*)&Bs[nt * 16 + ln16][kb];
            bfrag[nt].q[1] = *(const uint4*)&Bs[nt * 16 + ln16][16 + kb];
        }
#pragma unroll
        for (int nt = 0; nt < 4; ++nt) {
            acc[nt] = __builtin_amdgcn_wmma_f32_16x16x32_bf16(
                false, af.v, false, bfrag[nt].v, (short)0, acc[nt], false, false);
        }
        __syncthreads();
    }

    // Epilogue: C/D 16x16 f32 layout: VGPR v -> row v + 8*hseg, col = ln16
#pragma unroll
    for (int nt = 0; nt < 4; ++nt) {
        int col = bn + nt * 16 + ln16;
        float bv = bias ? bias[col] : 0.0f;
#pragma unroll
        for (int v = 0; v < 8; ++v) {
            int row = bm + wave * 16 + v + hseg * 8;
            C[(size_t)row * N + col] = acc[nt][v] + bv;
        }
    }
}

// ---------------- persistent recurrence kernel ----------------
#define NBLK 256   // persistent grid; 256 blocks x 8 waves = 2048 waves

__device__ __forceinline__ void grid_sync(unsigned* cnt, unsigned nb, unsigned& epoch) {
    __syncthreads();
    if (threadIdx.x == 0) {
        __threadfence();
        __hip_atomic_fetch_add(cnt, 1u, __ATOMIC_RELEASE, __HIP_MEMORY_SCOPE_AGENT);
        ++epoch;
        unsigned target = epoch * nb;
        while (__hip_atomic_load(cnt, __ATOMIC_ACQUIRE, __HIP_MEMORY_SCOPE_AGENT) < target)
            __builtin_amdgcn_s_sleep(1);
        __threadfence();
    }
    __syncthreads();
}

__global__ __launch_bounds__(256) void lstm_recurrence(
    const float* __restrict__ Xproj,   // [L][4H]  x-part of layer0 gates (+b0)
    const bf16_t* __restrict__ Wr0,    // [4H][H]  W_ih0[:, 4H:5H]  (h1 feedback)
    const bf16_t* __restrict__ Whh0,   // [4H][H]
    const bf16_t* __restrict__ Wih1,   // [4H][H]
    const bf16_t* __restrict__ Whh1,   // [4H][H]
    const float* __restrict__ b1,      // [4H]
    float* __restrict__ h0buf,         // [2][H] double buffer
    float* __restrict__ c0,            // [H]
    float* __restrict__ h1buf,         // [2][H] double buffer
    float* __restrict__ c1,            // [H]
    float* __restrict__ p1,            // [4H]  Whh1*h1_prev + b1
    bf16_t* __restrict__ outs,         // [L][H] bf16 copy of h1 (FC input)
    unsigned* __restrict__ cnt)
{
    const int tid  = threadIdx.x;
    const int wid  = blockIdx.x * 8 + (tid >> 5);   // 0..2047 (one wave per unit)
    const int lane = tid & 31;
    const unsigned nb = gridDim.x;
    unsigned epoch = 0;

    for (int t = 0; t < L; ++t) {
        const float* h0r = h0buf + (t & 1) * H;
        float*       h0w = h0buf + ((t + 1) & 1) * H;
        const float* h1r = h1buf + (t & 1) * H;
        float*       h1w = h1buf + ((t + 1) & 1) * H;

        // ---------------- Phase 1 ----------------
        if (wid < H) {
            // Layer-0 cell for element m: all 4 gate rows -> no extra sync needed.
            const int m = wid;
            float h1v[32], h0v[32];
            const float* hp1 = h1r + lane * 32;
            const float* hp0 = h0r + lane * 32;
#pragma unroll
            for (int i = 0; i < 32; ++i) { h1v[i] = hp1[i]; h0v[i] = hp0[i]; }

            float g[4];
#pragma unroll
            for (int q = 0; q < 4; ++q) {
                int row = m + q * H;
                const uint32_t* wr = (const uint32_t*)(Wr0  + (size_t)row * H) + lane * 16;
                const uint32_t* wh = (const uint32_t*)(Whh0 + (size_t)row * H) + lane * 16;
                float acc = 0.0f;
#pragma unroll
                for (int i = 0; i < 16; ++i) {
                    uint32_t a = wr[i], b = wh[i];
                    acc += bf2f_lo(a) * h1v[2 * i] + bf2f_hi(a) * h1v[2 * i + 1];
                    acc += bf2f_lo(b) * h0v[2 * i] + bf2f_hi(b) * h0v[2 * i + 1];
                }
#pragma unroll
                for (int s = 16; s > 0; s >>= 1) acc += __shfl_xor(acc, s, 32);
                g[q] = acc;
            }
            if (lane == 0) {
                const float* xp = Xproj + (size_t)t * G4H + m;
                float gi = g[0] + xp[0];
                float gf = g[1] + xp[1 * H];
                float gg = g[2] + xp[2 * H];
                float go = g[3] + xp[3 * H];
                float cn = sigf(gf) * c0[m] + sigf(gi) * tanhf(gg);
                c0[m]  = cn;
                h0w[m] = sigf(go) * tanhf(cn);
            }
        } else {
            // Concurrently: p1 = Whh1 * h1_prev + b1  (depends only on previous step)
            const int m = wid - H;
            float h1v[32];
            const float* hp1 = h1r + lane * 32;
#pragma unroll
            for (int i = 0; i < 32; ++i) h1v[i] = hp1[i];
#pragma unroll
            for (int q = 0; q < 4; ++q) {
                int row = m + q * H;
                const uint32_t* wh = (const uint32_t*)(Whh1 + (size_t)row * H) + lane * 16;
                float acc = 0.0f;
#pragma unroll
                for (int i = 0; i < 16; ++i) {
                    uint32_t b = wh[i];
                    acc += bf2f_lo(b) * h1v[2 * i] + bf2f_hi(b) * h1v[2 * i + 1];
                }
#pragma unroll
                for (int s = 16; s > 0; s >>= 1) acc += __shfl_xor(acc, s, 32);
                if (lane == 0) p1[row] = acc + b1[row];
            }
        }
        grid_sync(cnt, nb, epoch);

        // ---------------- Phase 2: layer-1 cell ----------------
        if (wid < H) {
            const int m = wid;
            float h0v[32];
            const float* hp0 = h0w + lane * 32;   // h0_new
#pragma unroll
            for (int i = 0; i < 32; ++i) h0v[i] = hp0[i];

            float g[4];
#pragma unroll
            for (int q = 0; q < 4; ++q) {
                int row = m + q * H;
                const uint32_t* wi = (const uint32_t*)(Wih1 + (size_t)row * H) + lane * 16;
                float acc = 0.0f;
#pragma unroll
                for (int i = 0; i < 16; ++i) {
                    uint32_t a = wi[i];
                    acc += bf2f_lo(a) * h0v[2 * i] + bf2f_hi(a) * h0v[2 * i + 1];
                }
#pragma unroll
                for (int s = 16; s > 0; s >>= 1) acc += __shfl_xor(acc, s, 32);
                g[q] = acc;
            }
            if (lane == 0) {
                float gi = g[0] + p1[m];
                float gf = g[1] + p1[m + 1 * H];
                float gg = g[2] + p1[m + 2 * H];
                float go = g[3] + p1[m + 3 * H];
                float cn = sigf(gf) * c1[m] + sigf(gi) * tanhf(gg);
                c1[m] = cn;
                float hn = sigf(go) * tanhf(cn);
                h1w[m] = hn;
                outs[(size_t)t * H + m] = f2bf(hn);
            }
        }
        grid_sync(cnt, nb, epoch);
    }
}

// ---------------- host side ----------------
extern "C" void kernel_launch(void* const* d_in, const int* in_sizes, int n_in,
                              void* d_out, int out_size, void* d_ws, size_t ws_size,
                              hipStream_t stream) {
    (void)in_sizes; (void)n_in; (void)out_size; (void)ws_size;

    const float* x     = (const float*)d_in[0];   // [1,L,2H]
    const float* hi    = (const float*)d_in[1];   // [1,L,2H]
    const float* W_ih0 = (const float*)d_in[2];   // [4H,5H]
    const float* W_hh0 = (const float*)d_in[3];   // [4H,H]
    const float* b0    = (const float*)d_in[4];   // [4H]
    const float* W_ih1 = (const float*)d_in[5];   // [4H,H]
    const float* W_hh1 = (const float*)d_in[6];   // [4H,H]
    const float* b1    = (const float*)d_in[7];   // [4H]
    const float* fc_W  = (const float*)d_in[8];   // [LBL,H]
    const float* fc_b  = (const float*)d_in[9];   // [LBL]
    float* out = (float*)d_out;                   // [1,L,LBL] f32

    // ---- workspace carving (all 256B aligned) ----
    uint8_t* w = (uint8_t*)d_ws;
    unsigned* cnt = (unsigned*)w;                  w += 256;
    float* state  = (float*)w;                     w += 6144 * sizeof(float); // h0buf[2H]+c0[H]+h1buf[2H]+c1[H]
    float* h0buf  = state;
    float* c0     = state + 2 * H;
    float* h1buf  = state + 3 * H;
    float* c1     = state + 5 * H;
    float* p1     = (float*)w;                     w += (size_t)G4H * sizeof(float);
    bf16_t* X4    = (bf16_t*)w;                    w += (size_t)L * G4H * 2;        // 32MB
    bf16_t* Wx    = (bf16_t*)w;                    w += (size_t)G4H * G4H * 2;      // 32MB
    bf16_t* Wr0   = (bf16_t*)w;                    w += (size_t)G4H * H * 2;        // 8MB
    bf16_t* Whh0b = (bf16_t*)w;                    w += (size_t)G4H * H * 2;        // 8MB
    bf16_t* Wih1b = (bf16_t*)w;                    w += (size_t)G4H * H * 2;        // 8MB
    bf16_t* Whh1b = (bf16_t*)w;                    w += (size_t)G4H * H * 2;        // 8MB
    bf16_t* fcWb  = (bf16_t*)w;                    w += (size_t)LBL * H * 2;        // 256KB
    float*  Xproj = (float*)w;                     w += (size_t)L * G4H * 4;        // 64MB
    bf16_t* outs  = (bf16_t*)w;                    w += (size_t)L * H * 2;          // 8MB

    // ---- prep: zero state + barrier counter, convert operands to bf16 ----
    k_init<<<24, 256, 0, stream>>>(state, 6144, cnt);
    k_pack_x<<<2048, 256, 0, stream>>>(x, hi, X4);
    k_cvt_bf16<<<2048, 256, 0, stream>>>(W_ih0,        5 * H, Wx,    G4H, G4H);  // x|hi part
    k_cvt_bf16<<<512,  256, 0, stream>>>(W_ih0 + 4 * H, 5 * H, Wr0,  G4H, H);    // h1-feedback part
    k_cvt_bf16<<<512,  256, 0, stream>>>(W_hh0,        H,     Whh0b, G4H, H);
    k_cvt_bf16<<<512,  256, 0, stream>>>(W_ih1,        H,     Wih1b, G4H, H);
    k_cvt_bf16<<<512,  256, 0, stream>>>(W_hh1,        H,     Whh1b, G4H, H);
    k_cvt_bf16<<<64,   256, 0, stream>>>(fc_W,         H,     fcWb,  LBL, H);

    // ---- big time-parallel GEMM: Xproj = X4 @ Wx^T + b0  (4096^3, WMMA bf16) ----
    {
        dim3 grid(G4H / BN, L / BM);   // (64, 32)
        gemm_bf16_wmma<<<grid, 256, 0, stream>>>(X4, Wx, b0, Xproj, L, G4H, G4H);
    }

    // ---- sequential recurrence: persistent kernel, 2 grid syncs/step ----
    lstm_recurrence<<<NBLK, 256, 0, stream>>>(Xproj, Wr0, Whh0b, Wih1b, Whh1b, b1,
                                              h0buf, c0, h1buf, c1, p1, outs, cnt);

    // ---- FC epilogue: out = outs @ fc_W^T + fc_b  (WMMA) ----
    {
        dim3 grid(LBL / BN, L / BM);   // (2, 32)
        gemm_bf16_wmma<<<grid, 256, 0, stream>>>(outs, fcWb, fc_b, out, L, LBL, H);
    }
}
